// GATNet_47330539602645
// MI455X (gfx1250) — matrix operand order repaired and checked by reading, not compiled, not run on hardware
//
#include <hip/hip_runtime.h>
#include <hip/hip_bf16.h>

// ---------------- problem constants ----------------
#define NND   50000
#define NEDG  1600000
#define ETOT  1650000          // edges + self loops
#define FIN   256
#define FMID  64
#define FOUT  128
#define H1N   8
#define C1N   8
#define NEGS  0.2f

typedef __attribute__((ext_vector_type(16))) __bf16 v16bf;
typedef __attribute__((ext_vector_type(8)))  __bf16 v8bf;
typedef __attribute__((ext_vector_type(8)))  float  v8f;

// ---------------- helpers ----------------
__device__ __forceinline__ unsigned f2ord(float f) {
    unsigned u = __float_as_uint(f);
    return (u & 0x80000000u) ? ~u : (u | 0x80000000u);
}
__device__ __forceinline__ float ord2f(unsigned u) {
    return (u & 0x80000000u) ? __uint_as_float(u & 0x7fffffffu)
                             : __uint_as_float(~u);
}
__device__ __forceinline__ float lrelu(float x) { return x > 0.f ? x : NEGS * x; }

__device__ __forceinline__ void edge_pair(const long long* ei, int e, int& s, int& d) {
    if (e < NEDG) { s = (int)ei[e]; d = (int)ei[NEDG + e]; }
    else          { s = d = e - NEDG; }                     // self loop
}

// ---------------- small utility kernels ----------------
__global__ void k_zero_f32(float* p, int n) {
    int i = blockIdx.x * blockDim.x + threadIdx.x;
    if (i < n) p[i] = 0.f;
}
__global__ void k_cast_bf16(const float* __restrict__ in, __bf16* __restrict__ out, int n) {
    int i = blockIdx.x * blockDim.x + threadIdx.x;
    if (i < n) out[i] = (__bf16)in[i];
}
// W[K][Nc] row-major (f32) -> Wt[Nc][K] bf16 (so B columns become contiguous)
__global__ void k_transpose_cast(const float* __restrict__ W, __bf16* __restrict__ Wt,
                                 int K, int Nc) {
    int i = blockIdx.x * blockDim.x + threadIdx.x;
    if (i >= K * Nc) return;
    int k = i / Nc, n = i % Nc;
    Wt[(size_t)n * K + k] = (__bf16)W[(size_t)k * Nc + n];
}

// ---------------- WMMA GEMM: C[M,N] = A[M,K](bf16) x Bt[N,K](bf16), f32 accum -----
// One wave -> 16 rows x (16*NT) cols.  M % 16 == 0, K % 32 == 0, N % (16*NT) == 0.
template<int K, int NT>
__global__ void k_gemm_bf16_wmma(const __bf16* __restrict__ A,
                                 const __bf16* __restrict__ Bt,
                                 float* __restrict__ C,
                                 int N, int totalWaves) {
    const int wid = blockIdx.x * (blockDim.x >> 5) + (threadIdx.x >> 5);
    if (wid >= totalWaves) return;
    const int lane   = threadIdx.x & 31;
    const int wavesN = N / (16 * NT);
    const int mt     = wid / wavesN;
    const int nt     = wid % wavesN;
    const int row    = mt * 16 + (lane & 15);
    const int colb   = nt * 16 * NT + (lane & 15);
    const int hi     = lane >> 4;            // which half-wave

    v8f acc[NT] = {};
    for (int k0 = 0; k0 < K; k0 += 32) {
        // A fragment, contiguous K order (matches B):
        // lanes 0-15: elems 0..15 = K k0..k0+15
        // lanes16-31: elems 0..15 = K k0+16..k0+31
        const int ka = k0 + hi * 16;
        const v16bf af = *(const v16bf*)(A + (size_t)row * K + ka);

        // B fragment: lane holds one column, 16 consecutive K values
        const int kb = k0 + hi * 16;
#pragma unroll
        for (int t = 0; t < NT; ++t) {
            const v16bf bf = *(const v16bf*)(Bt + (size_t)(colb + t * 16) * K + kb);
            acc[t] = __builtin_amdgcn_wmma_f32_16x16x32_bf16(
                         false, af, false, bf, (short)0, acc[t], false, false);
        }
    }
    // C/D layout: VGPR r -> M = r + 8*hi, N = lane&15 (+tile offset)
    const int crow = mt * 16 + hi * 8;
#pragma unroll
    for (int t = 0; t < NT; ++t) {
        const int ccol = colb + t * 16;
#pragma unroll
        for (int r = 0; r < 8; ++r)
            C[(size_t)(crow + r) * N + ccol] = acc[t][r];
    }
}

// ---------------- layer 1 (H=8, C=8) ----------------
__global__ void k_alpha1(const float* __restrict__ h1,
                         const float* __restrict__ as, const float* __restrict__ ad,
                         float* __restrict__ als, float* __restrict__ ald) {
    int i = blockIdx.x * blockDim.x + threadIdx.x;           // node*8 + head
    if (i >= NND * H1N) return;
    int n = i >> 3, h = i & 7;
    const float* base = h1 + (size_t)n * FMID + h * C1N;
    float ss = 0.f, sd = 0.f;
#pragma unroll
    for (int c = 0; c < C1N; ++c) {
        float v = base[c];
        ss += v * as[h * C1N + c];
        sd += v * ad[h * C1N + c];
    }
    als[i] = ss; ald[i] = sd;
}

__global__ void k_edge_max1(const long long* __restrict__ ei,
                            const float* __restrict__ als, const float* __restrict__ ald,
                            unsigned* __restrict__ m) {
    int e = blockIdx.x * blockDim.x + threadIdx.x;
    if (e >= ETOT) return;
    int s, d; edge_pair(ei, e, s, d);
#pragma unroll
    for (int h = 0; h < H1N; ++h) {
        float v = lrelu(als[s * H1N + h] + ald[d * H1N + h]);
        atomicMax(m + d * H1N + h, f2ord(v));
    }
}

__global__ void k_edge_acc1(const long long* __restrict__ ei,
                            const float* __restrict__ als, const float* __restrict__ ald,
                            const unsigned* __restrict__ m,
                            const float* __restrict__ h1,
                            float* __restrict__ denom, float* __restrict__ msg) {
    int i = blockIdx.x * blockDim.x + threadIdx.x;           // edge*8 + head
    if (i >= ETOT * H1N) return;
    int e = i >> 3, h = i & 7;
    int s, d; edge_pair(ei, e, s, d);
    float v = lrelu(als[s * H1N + h] + ald[d * H1N + h]);
    float p = __expf(v - ord2f(m[d * H1N + h]));
    atomicAdd(denom + d * H1N + h, p);
    const float4 hv0 = *(const float4*)(h1 + (size_t)s * FMID + h * C1N);
    const float4 hv1 = *(const float4*)(h1 + (size_t)s * FMID + h * C1N + 4);
    float* mb = msg + (size_t)d * FMID + h * C1N;
    atomicAdd(mb + 0, p * hv0.x); atomicAdd(mb + 1, p * hv0.y);
    atomicAdd(mb + 2, p * hv0.z); atomicAdd(mb + 3, p * hv0.w);
    atomicAdd(mb + 4, p * hv1.x); atomicAdd(mb + 5, p * hv1.y);
    atomicAdd(mb + 6, p * hv1.z); atomicAdd(mb + 7, p * hv1.w);
}

// normalize + bias + ELU, emit bf16 for GEMM2
__global__ void k_final1(const float* __restrict__ msg, const float* __restrict__ denom,
                         const float* __restrict__ b1, __bf16* __restrict__ hmid) {
    int i = blockIdx.x * blockDim.x + threadIdx.x;           // node*64 + f
    if (i >= NND * FMID) return;
    int n = i >> 6, f = i & 63, h = f >> 3;
    float v = msg[i] / (denom[n * H1N + h] + 1e-16f) + b1[f];
    v = v > 0.f ? v : __expf(v) - 1.f;                       // ELU
    hmid[i] = (__bf16)v;
}

// ---------------- layer 2 (H=1, C=128) ----------------
__global__ void k_alpha2(const float* __restrict__ h2,
                         const float* __restrict__ as, const float* __restrict__ ad,
                         float* __restrict__ als, float* __restrict__ ald) {
    int n    = blockIdx.x * (blockDim.x >> 5) + (threadIdx.x >> 5);
    int lane = threadIdx.x & 31;
    if (n >= NND) return;
    float ss = 0.f, sd = 0.f;
#pragma unroll
    for (int j = 0; j < 4; ++j) {
        int c = lane + 32 * j;
        float v = h2[(size_t)n * FOUT + c];
        ss += v * as[c]; sd += v * ad[c];
    }
#pragma unroll
    for (int off = 16; off > 0; off >>= 1) {
        ss += __shfl_down(ss, off, 32);
        sd += __shfl_down(sd, off, 32);
    }
    if (lane == 0) { als[n] = ss; ald[n] = sd; }
}

__global__ void k_edge_max2(const long long* __restrict__ ei,
                            const float* __restrict__ als, const float* __restrict__ ald,
                            unsigned* __restrict__ m) {
    int e = blockIdx.x * blockDim.x + threadIdx.x;
    if (e >= ETOT) return;
    int s, d; edge_pair(ei, e, s, d);
    atomicMax(m + d, f2ord(lrelu(als[s] + ald[d])));
}

__global__ void k_edge_acc2(const long long* __restrict__ ei,
                            const float* __restrict__ als, const float* __restrict__ ald,
                            const unsigned* __restrict__ m,
                            const float* __restrict__ h2,
                            float* __restrict__ denom, float* __restrict__ msg) {
    int e    = blockIdx.x * (blockDim.x >> 5) + (threadIdx.x >> 5);  // wave per edge
    int lane = threadIdx.x & 31;
    if (e >= ETOT) return;
    int s, d; edge_pair(ei, e, s, d);
    float p = __expf(lrelu(als[s] + ald[d]) - ord2f(m[d]));
    if (lane == 0) atomicAdd(denom + d, p);
#pragma unroll
    for (int j = 0; j < 4; ++j) {
        int c = lane + 32 * j;
        atomicAdd(msg + (size_t)d * FOUT + c, p * h2[(size_t)s * FOUT + c]);
    }
}

__global__ void k_final2(const float* __restrict__ msg, const float* __restrict__ denom,
                         const float* __restrict__ b2, float* __restrict__ out) {
    int i = blockIdx.x * blockDim.x + threadIdx.x;           // node*128 + f
    if (i >= NND * FOUT) return;
    int n = i >> 7, f = i & 127;
    out[i] = msg[i] / (denom[n] + 1e-16f) + b2[f];
}

// ---------------- host launcher ----------------
extern "C" void kernel_launch(void* const* d_in, const int* in_sizes, int n_in,
                              void* d_out, int out_size, void* d_ws, size_t ws_size,
                              hipStream_t stream) {
    const float*     x   = (const float*)d_in[0];
    const long long* ei  = (const long long*)d_in[1];
    const float*     w1  = (const float*)d_in[2];
    const float*     as1 = (const float*)d_in[3];
    const float*     ad1 = (const float*)d_in[4];
    const float*     b1  = (const float*)d_in[5];
    const float*     w2  = (const float*)d_in[6];
    const float*     as2 = (const float*)d_in[7];
    const float*     ad2 = (const float*)d_in[8];
    const float*     b2  = (const float*)d_in[9];
    float*           out = (float*)d_out;

    char* ws = (char*)d_ws;
    size_t off = 0;
    auto take = [&](size_t bytes) { off = (off + 255) & ~(size_t)255;
                                    size_t o = off; off += bytes; return o; };

    __bf16*   xbf   = (__bf16*)(ws + take((size_t)NND * FIN * 2));
    __bf16*   w1t   = (__bf16*)(ws + take((size_t)FMID * FIN * 2));
    __bf16*   w2t   = (__bf16*)(ws + take((size_t)FOUT * FMID * 2));
    float*    h1    = (float*) (ws + take((size_t)NND * FMID * 4));
    float*    als1  = (float*) (ws + take((size_t)NND * H1N * 4));
    float*    ald1  = (float*) (ws + take((size_t)NND * H1N * 4));
    __bf16*   hmid  = (__bf16*)(ws + take((size_t)NND * FMID * 2));
    float*    h2    = (float*) (ws + take((size_t)NND * FOUT * 4));
    float*    als2  = (float*) (ws + take((size_t)NND * 4));
    float*    ald2  = (float*) (ws + take((size_t)NND * 4));
    unsigned* m1    = (unsigned*)(ws + take((size_t)NND * H1N * 4));
    float*    den1  = (float*) (ws + take((size_t)NND * H1N * 4));
    float*    msg1  = (float*) (ws + take((size_t)NND * FMID * 4));
    unsigned* m2    = (unsigned*)(ws + take((size_t)NND * 4));
    float*    den2  = (float*) (ws + take((size_t)NND * 4));
    float*    msg2  = (float*) (ws + take((size_t)NND * FOUT * 4));

    const int T = 256;
    auto blk = [](long long n, int t) { return (int)((n + t - 1) / t); };

    // zero the accumulators (fresh every call — deterministic)
    k_zero_f32<<<blk((long long)NND * H1N, T), T, 0, stream>>>((float*)m1, NND * H1N);
    k_zero_f32<<<blk((long long)NND * H1N, T), T, 0, stream>>>(den1, NND * H1N);
    k_zero_f32<<<blk((long long)NND * FMID, T), T, 0, stream>>>(msg1, NND * FMID);
    k_zero_f32<<<blk(NND, T), T, 0, stream>>>((float*)m2, NND);
    k_zero_f32<<<blk(NND, T), T, 0, stream>>>(den2, NND);
    k_zero_f32<<<blk((long long)NND * FOUT, T), T, 0, stream>>>(msg2, NND * FOUT);

    // precision cast + weight transpose for WMMA
    k_cast_bf16<<<blk((long long)NND * FIN, T), T, 0, stream>>>(x, xbf, NND * FIN);
    k_transpose_cast<<<blk((long long)FIN * FMID, T), T, 0, stream>>>(w1, w1t, FIN, FMID);
    k_transpose_cast<<<blk((long long)FMID * FOUT, T), T, 0, stream>>>(w2, w2t, FMID, FOUT);

    // GEMM1: h1 = x @ W1   (wave = 16 rows x 64 cols, K=256)
    {
        int waves = (NND / 16) * (FMID / 64);
        k_gemm_bf16_wmma<FIN, 4><<<blk((long long)waves * 32, T), T, 0, stream>>>(
            xbf, w1t, h1, FMID, waves);
    }

    // layer-1 attention
    k_alpha1<<<blk((long long)NND * H1N, T), T, 0, stream>>>(h1, as1, ad1, als1, ald1);
    k_edge_max1<<<blk(ETOT, T), T, 0, stream>>>(ei, als1, ald1, m1);
    k_edge_acc1<<<blk((long long)ETOT * H1N, T), T, 0, stream>>>(ei, als1, ald1, m1, h1,
                                                                 den1, msg1);
    k_final1<<<blk((long long)NND * FMID, T), T, 0, stream>>>(msg1, den1, b1, hmid);

    // GEMM2: h2 = hmid @ W2   (K=64, N=128 -> 2 waves per 16-row strip)
    {
        int waves = (NND / 16) * (FOUT / 64);
        k_gemm_bf16_wmma<FMID, 4><<<blk((long long)waves * 32, T), T, 0, stream>>>(
            hmid, w2t, h2, FOUT, waves);
    }

    // layer-2 attention
    k_alpha2<<<blk((long long)NND * 32, T), T, 0, stream>>>(h2, as2, ad2, als2, ald2);
    k_edge_max2<<<blk(ETOT, T), T, 0, stream>>>(ei, als2, ald2, m2);
    k_edge_acc2<<<blk((long long)ETOT * 32, T), T, 0, stream>>>(ei, als2, ald2, m2, h2,
                                                                den2, msg2);
    k_final2<<<blk((long long)NND * FOUT, T), T, 0, stream>>>(msg2, den2, b2, out);
}
